// ClusterLoss_56573309224210
// MI455X (gfx1250) — matrix operand order repaired and checked by reading, not compiled, not run on hardware
//
#include <hip/hip_runtime.h>
#include <math.h>

typedef __attribute__((ext_vector_type(16))) _Float16 v16h;
typedef __attribute__((ext_vector_type(8)))  _Float16 v8h;
typedef __attribute__((ext_vector_type(4)))  _Float16 v4h;
typedef __attribute__((ext_vector_type(8)))  float    v8f;

#define DIM 128
#define EPSN 1e-12f

// ---------------------------------------------------------------------------
// Kernel 1: normalize centers -> f16, store ||c|| (post-eps scale) and ||c||^2
// grid = K blocks of 128 threads (one block per center row)
// ---------------------------------------------------------------------------
__global__ void prep_centers(const float* __restrict__ centers,
                             _Float16* __restrict__ chat,
                             float* __restrict__ cnorm,
                             float* __restrict__ cnsq) {
    const int k = blockIdx.x;
    const int t = threadIdx.x;           // 0..127
    float x = centers[(size_t)k * DIM + t];
    float s = x * x;
    #pragma unroll
    for (int m = 16; m >= 1; m >>= 1) s += __shfl_xor(s, m, 32);
    __shared__ float red[4];
    if ((t & 31) == 0) red[t >> 5] = s;
    __syncthreads();
    float tot   = red[0] + red[1] + red[2] + red[3];
    float scale = fmaxf(sqrtf(tot), EPSN);   // c = chat * scale
    chat[(size_t)k * DIM + t] = (_Float16)(x / scale);
    if (t == 0) { cnorm[k] = scale; cnsq[k] = tot; }
}

// ---------------------------------------------------------------------------
// Kernel 2: normalize features -> f16. One wave per row (32 lanes x 4 elems).
// blockDim = 256 (8 waves -> 8 rows per block)
// ---------------------------------------------------------------------------
__global__ void prep_features(const float* __restrict__ feat,
                              _Float16* __restrict__ fhat, int N) {
    const int row  = blockIdx.x * 8 + (threadIdx.x >> 5);
    const int lane = threadIdx.x & 31;
    if (row >= N) return;
    const float4 v = ((const float4*)(feat + (size_t)row * DIM))[lane];
    float s = v.x * v.x + v.y * v.y + v.z * v.z + v.w * v.w;
    #pragma unroll
    for (int m = 16; m >= 1; m >>= 1) s += __shfl_xor(s, m, 32);
    const float r = 1.0f / fmaxf(sqrtf(s), EPSN);
    v4h o;
    o[0] = (_Float16)(v.x * r);
    o[1] = (_Float16)(v.y * r);
    o[2] = (_Float16)(v.z * r);
    o[3] = (_Float16)(v.w * r);
    ((v4h*)(fhat + (size_t)row * DIM))[lane] = o;
}

__global__ void zero_out(float* out) {
    if (threadIdx.x == 0 && blockIdx.x == 0) out[0] = 0.0f;
}

// ---------------------------------------------------------------------------
// Kernel 3: WMMA GEMM (f16 in, f32 acc) + running argmax + closed-form loss.
// Each wave handles M=32 rows (two 16x16 C tiles sharing B fragments),
// loops over all K=1024 centers in 16-column tiles, D=128 via 4x K=32 WMMA.
// blockDim = 128 (4 waves); grid = N/128 blocks.
// ---------------------------------------------------------------------------
__global__ void __launch_bounds__(128)
cluster_main(const _Float16* __restrict__ fhat,
             const _Float16* __restrict__ chat,
             const float* __restrict__ cnorm,
             const float* __restrict__ cnsq,
             float* __restrict__ out, int N, int K) {
    const int lane    = threadIdx.x & 31;
    const int gw      = blockIdx.x * 4 + (threadIdx.x >> 5);
    const int rowbase = gw * 32;
    const int half    = lane >> 4;   // 0: lanes 0-15, 1: lanes 16-31
    const int nn      = lane & 15;

    // ---- Load A fragments once: 16-bit A 16x32, contiguous k order ----
    // lane (half,nn): elements [0..15] = K = half*16 + i (contiguous, matches B)
    v16h afrag[2][4];
    #pragma unroll
    for (int t = 0; t < 2; ++t) {
        const _Float16* ar = fhat + (size_t)(rowbase + t * 16 + nn) * DIM;
        #pragma unroll
        for (int dc = 0; dc < 4; ++dc) {
            v8h lo = *(const v8h*)(ar + dc * 32 + half * 16);
            v8h hi = *(const v8h*)(ar + dc * 32 + half * 16 + 8);
            v16h a;
            #pragma unroll
            for (int i = 0; i < 8; ++i) { a[i] = lo[i]; a[i + 8] = hi[i]; }
            afrag[t][dc] = a;
        }
    }

    float bestv[2][8];
    int   besti[2][8];
    #pragma unroll
    for (int t = 0; t < 2; ++t)
        #pragma unroll
        for (int j = 0; j < 8; ++j) { bestv[t][j] = -3.0e38f; besti[t][j] = 0; }

    // ---- Main loop over 16-center column tiles ----
    for (int kt = 0; kt < K / 16; ++kt) {
        // B fragment, 16-bit B 32x16, contiguous k order: lane (half,nn) holds
        // column-center (kt*16+nn), K = half*16 + 0..15 (contiguous in row-major chat)
        const _Float16* cr = chat + (size_t)(kt * 16 + nn) * DIM + half * 16;
        v8f acc0 = {};
        v8f acc1 = {};
        #pragma unroll
        for (int dc = 0; dc < 4; ++dc) {
            v16h b = *(const v16h*)(cr + dc * 32);
            acc0 = __builtin_amdgcn_wmma_f32_16x16x32_f16(
                false, afrag[0][dc], false, b, (short)0, acc0, false, false);
            acc1 = __builtin_amdgcn_wmma_f32_16x16x32_f16(
                false, afrag[1][dc], false, b, (short)0, acc1, false, false);
        }
        // C layout: lane column N = nn, VGPR j -> row (half*8 + j) of the tile
        const int col = kt * 16 + nn;
        #pragma unroll
        for (int j = 0; j < 8; ++j) {
            float v0 = acc0[j];
            if (v0 > bestv[0][j]) { bestv[0][j] = v0; besti[0][j] = col; }
            float v1 = acc1[j];
            if (v1 > bestv[1][j]) { bestv[1][j] = v1; besti[1][j] = col; }
        }
    }

    // ---- Argmax reduction across the 16 lanes of each half-group, epilogue ----
    float lsum = 0.0f;
    #pragma unroll
    for (int t = 0; t < 2; ++t) {
        #pragma unroll
        for (int j = 0; j < 8; ++j) {
            float v = bestv[t][j];
            int   i = besti[t][j];
            #pragma unroll
            for (int m = 1; m < 16; m <<= 1) {
                float ov = __shfl_xor(v, m, 32);
                int   oi = __shfl_xor(i, m, 32);
                if (ov > v || (ov == v && oi < i)) { v = ov; i = oi; }
            }
            if (nn == 0) {
                // row = rowbase + t*16 + half*8 + j ; ||f||==1 after normalize
                lsum += 1.0f - 2.0f * v * cnorm[i] + cnsq[i];
            }
        }
    }
    // lanes 0 and 16 hold sums of 16 rows each; fold and accumulate
    lsum += __shfl_xor(lsum, 16, 32);
    if (lane == 0) atomicAdd(out, lsum * (1.0f / (float)N));
}

// ---------------------------------------------------------------------------
extern "C" void kernel_launch(void* const* d_in, const int* in_sizes, int n_in,
                              void* d_out, int out_size, void* d_ws, size_t ws_size,
                              hipStream_t stream) {
    const float* features = (const float*)d_in[0];
    const float* centers  = (const float*)d_in[1];
    float* out = (float*)d_out;

    const int N = in_sizes[0] / DIM;   // 131072
    const int K = in_sizes[1] / DIM;   // 1024

    char* ws = (char*)d_ws;
    _Float16* fhat = (_Float16*)ws;                       // N*128*2 = 32 MB
    size_t fb = (size_t)N * DIM * sizeof(_Float16);
    _Float16* chat = (_Float16*)(ws + fb);                // K*128*2 = 256 KB (32B-aligned)
    size_t cb = (size_t)K * DIM * sizeof(_Float16);
    float* cnorm = (float*)(ws + fb + cb);                // K floats
    float* cnsq  = cnorm + K;                             // K floats

    prep_centers <<<K,          128, 0, stream>>>(centers, chat, cnorm, cnsq);
    prep_features<<<(N + 7)/8,  256, 0, stream>>>(features, fhat, N);
    zero_out     <<<1,           64, 0, stream>>>(out);
    cluster_main <<<N / 128,    128, 0, stream>>>(fhat, chat, cnorm, cnsq, out, N, K);
}
